// Attention_51316269253426
// MI455X (gfx1250) — matrix operand order repaired and checked by reading, not compiled, not run on hardware
//
#include <hip/hip_runtime.h>

typedef __bf16 bf16_t;
typedef __attribute__((ext_vector_type(16))) __bf16 v16bf;
typedef __attribute__((ext_vector_type(8)))  __bf16 v8bf;
typedef __attribute__((ext_vector_type(8)))  float  v8f;
typedef __attribute__((ext_vector_type(4)))  unsigned int v4u;
typedef __attribute__((ext_vector_type(8)))  int v8i;
typedef __attribute__((ext_vector_type(4)))  int v4i;

// ---------------------------------------------------------------------------
// WMMA fragment loaders (wave32, 16x16x32 bf16)
// A 16x32: lane(0-15)=row M, contiguous 16-element K half per lane group.
// B 32x16: lane(0-15)=col N, contiguous 16-element K half per lane group.
// ---------------------------------------------------------------------------
__device__ __forceinline__ v16bf load_fragA(const bf16_t* base, uint32_t off) {
  v8bf lo = *(const v8bf*)(base + off);
  v8bf hi = *(const v8bf*)(base + off + 8);
  v16bf f;
#pragma unroll
  for (int i = 0; i < 8; ++i) { f[i] = lo[i]; f[i + 8] = hi[i]; }
  return f;
}

__device__ __forceinline__ v16bf load_fragB(const bf16_t* base, uint32_t off) {
  v8bf lo = *(const v8bf*)(base + off);
  v8bf hi = *(const v8bf*)(base + off + 8);
  v16bf f;
#pragma unroll
  for (int i = 0; i < 8; ++i) { f[i] = lo[i]; f[i + 8] = hi[i]; }
  return f;
}

__device__ __forceinline__ v16bf load_fragA_lds(const bf16_t* p) {
  v8bf lo = *(const v8bf*)(p);
  v8bf hi = *(const v8bf*)(p + 8);
  v16bf f;
#pragma unroll
  for (int i = 0; i < 8; ++i) { f[i] = lo[i]; f[i + 8] = hi[i]; }
  return f;
}

__device__ __forceinline__ v16bf load_fragB_lds(const bf16_t* p) {
  v8bf lo = *(const v8bf*)(p);
  v8bf hi = *(const v8bf*)(p + 8);
  v16bf f;
#pragma unroll
  for (int i = 0; i < 8; ++i) { f[i] = lo[i]; f[i + 8] = hi[i]; }
  return f;
}

__device__ __forceinline__ v8f wmma_bf16(v16bf a, v16bf b, v8f c) {
  return __builtin_amdgcn_wmma_f32_16x16x32_bf16(false, a, false, b, (short)0, c,
                                                 false, false);
}

// ---------------------------------------------------------------------------
// TDM: async-load a 2D bf16 tile [64 rows x 32 k-elems] of Bt[N,K] into
// LDS at byte offset lds_off. D# layout per CDNA5 ISA 8.3/8.4:
//  g0: [1:0]=count=1, [63:32]=lds_addr, [120:64]=global_addr, [127:126]=type=2
//  g1: [17:16]=data_size(1=2B), [79:48]=tensor_dim0, [111:80]=tensor_dim1,
//      [127:112]=tile_dim0, [143:128]=tile_dim1, [207:160]=dim0_stride
// clang-23 toolchain: 6-arg builtin (g0, g1, g2, g3, extra group, cpol).
// ---------------------------------------------------------------------------
__device__ __forceinline__ void tdm_load_Btile(const bf16_t* Bt, int n0, int k0,
                                               int K, int N, uint32_t lds_off) {
  const uint64_t ga = (uint64_t)(uintptr_t)(Bt + (size_t)n0 * K + k0);
  v4u g0;
  g0[0] = 1u;                                             // count=1, user mode
  g0[1] = lds_off;                                        // lds_addr (bytes)
  g0[2] = (uint32_t)ga;                                   // global_addr[31:0]
  g0[3] = (uint32_t)((ga >> 32) & 0x01FFFFFFu) | (2u << 30); // addr[56:32]|type=2
  v8i g1;
  g1[0] = (int)(1u << 16);                                // data_size = 2 bytes
  g1[1] = (int)(((uint32_t)K & 0xFFFFu) << 16);           // tensor_dim0[15:0]
  g1[2] = (int)((((uint32_t)K >> 16) & 0xFFFFu) |
                (((uint32_t)N & 0xFFFFu) << 16));         // dim0[31:16]|dim1[15:0]
  g1[3] = (int)((((uint32_t)N >> 16) & 0xFFFFu) |
                (32u << 16));                             // dim1[31:16]|tile_dim0=32
  g1[4] = (int)64u;                                       // tile_dim1=64, tile_dim2=0
  g1[5] = (int)(uint32_t)K;                               // dim0_stride[31:0]
  g1[6] = 0;                                              // stride hi / dim1_stride lo
  g1[7] = 0;
  v4i g2; g2[0] = 0; g2[1] = 0; g2[2] = 0; g2[3] = 0;
  v4i g3; g3[0] = 0; g3[1] = 0; g3[2] = 0; g3[3] = 0;
  v8i g4; g4[0] = 0; g4[1] = 0; g4[2] = 0; g4[3] = 0;
  g4[4] = 0; g4[5] = 0; g4[6] = 0; g4[7] = 0;
  __builtin_amdgcn_tensor_load_to_lds(g0, g1, g2, g3, g4, 0);
}

// ---------------------------------------------------------------------------
// Elementwise fp32 -> bf16
// ---------------------------------------------------------------------------
__global__ void k_f32_to_bf16(const float* __restrict__ in, bf16_t* __restrict__ out,
                              size_t n) {
  size_t i = (size_t)blockIdx.x * blockDim.x + threadIdx.x;
  if (i < n) out[i] = (bf16_t)in[i];
}

// W[K,N] fp32 -> Wt[N,K] bf16
__global__ void k_transpose_bf16(const float* __restrict__ W, bf16_t* __restrict__ Wt,
                                 int Kdim, int Ndim) {
  size_t i = (size_t)blockIdx.x * blockDim.x + threadIdx.x;
  if (i >= (size_t)Kdim * Ndim) return;
  int k = (int)(i % Kdim);
  int n = (int)(i / Kdim);
  Wt[(size_t)n * Kdim + k] = (bf16_t)W[(size_t)k * Ndim + n];
}

// ---------------------------------------------------------------------------
// GEMM: C[M,N] = A[M,K] * Bt[N,K]^T   (A,Bt bf16 row-major, f32 accumulate)
// block = 256 threads (8 waves). Wave w -> rows [blockIdx.y*256 + w*32, +32),
// cols [blockIdx.x*64, +64): 2x4 = 8 WMMA tiles per k-step.
// B k-slices (64x32) are DMA'd into double-buffered LDS by the TDM (wave 0),
// overlapped with WMMA; A fragments are register double-buffered.
// ---------------------------------------------------------------------------
__global__ void __launch_bounds__(256, 1)
k_gemm_bf16(const bf16_t* __restrict__ A, const bf16_t* __restrict__ Bt,
            float* __restrict__ Cf, bf16_t* __restrict__ Cb,
            const float* __restrict__ bias, int M, int N, int K) {
  __shared__ bf16_t Blds[2][64 * 32];   // 2 x 4KB double buffer

  const int lane = threadIdx.x & 31;
  const int wave = threadIdx.x >> 5;
  const int half = lane >> 4;
  const int l16  = lane & 15;
  const int row0 = blockIdx.y * 256 + wave * 32;
  const int col0 = blockIdx.x * 64;

  v8f acc[2][4];
#pragma unroll
  for (int mi = 0; mi < 2; ++mi)
#pragma unroll
    for (int t = 0; t < 4; ++t)
#pragma unroll
      for (int r = 0; r < 8; ++r) acc[mi][t][r] = 0.0f;

  // 32-bit element offsets (uniform base A stays in SGPRs)
  const uint32_t aoff0 = (uint32_t)(row0 + l16) * (uint32_t)K + half * 16;
  const uint32_t aoff1 = aoff0 + 16u * (uint32_t)K;

  // prologue: TDM brings B k-slice 0 into buffer 0; A frags for k=0 in regs
  if (wave == 0) tdm_load_Btile(Bt, col0, 0, K, N, 0u);
  v16bf a_cur[2];
  a_cur[0] = load_fragA(A, aoff0);
  a_cur[1] = load_fragA(A, aoff1);

  int it = 0;
#pragma unroll 2
  for (int k0 = 0; k0 < K; k0 += 32, ++it) {
    // current buffer ready: issuing wave drains TENSORcnt, then block syncs.
    if (wave == 0) __builtin_amdgcn_s_wait_tensorcnt(0);
    __syncthreads();

    // kick off next k-slice DMA into the other buffer (overlaps WMMAs below)
    if (wave == 0 && k0 + 32 < K)
      tdm_load_Btile(Bt, col0, k0 + 32, K, N, (uint32_t)(((it + 1) & 1) * 4096));

    // prefetch + issue next A fragment loads before this step's WMMAs
    const uint32_t k1 = (k0 + 32 < K) ? (uint32_t)(k0 + 32) : 0u; // dummy at end
    __builtin_prefetch(A + aoff0 + k0 + 64, 0, 3);
    __builtin_prefetch(A + aoff1 + k0 + 64, 0, 3);
    v16bf a_nxt[2];
    a_nxt[0] = load_fragA(A, aoff0 + k1);
    a_nxt[1] = load_fragA(A, aoff1 + k1);

    // B fragments from LDS (shared by all 8 waves)
    const bf16_t* bbuf = &Blds[it & 1][0];
    v16bf b[4];
#pragma unroll
    for (int t = 0; t < 4; ++t)
      b[t] = load_fragB_lds(bbuf + (t * 16 + l16) * 32 + half * 16);

#pragma unroll
    for (int t = 0; t < 4; ++t) {
      acc[0][t] = wmma_bf16(a_cur[0], b[t], acc[0][t]);
      acc[1][t] = wmma_bf16(a_cur[1], b[t], acc[1][t]);
    }

    a_cur[0] = a_nxt[0];
    a_cur[1] = a_nxt[1];
  }

#pragma unroll
  for (int mi = 0; mi < 2; ++mi) {
#pragma unroll
    for (int t = 0; t < 4; ++t) {
      const int col = col0 + t * 16 + l16;
#pragma unroll
      for (int r = 0; r < 8; ++r) {
        const int row = row0 + mi * 16 + r + 8 * half;
        float v = acc[mi][t][r];
        if (Cf) Cf[(size_t)row * N + col] = v + (bias ? bias[col] : 0.0f);
        else    Cb[(size_t)row * N + col] = (bf16_t)v;
      }
    }
  }
}

// ---------------------------------------------------------------------------
// Repack qkv bf16 [b, n, 3*1024] -> Q[b,h,n,64], K[b,h,n,64], Vt[b,h,64,n]
// ---------------------------------------------------------------------------
__global__ void k_repack_qkv(const bf16_t* __restrict__ qkv, bf16_t* __restrict__ Q,
                             bf16_t* __restrict__ Km, bf16_t* __restrict__ Vt) {
  const int SEQ = 2048;
  size_t idx = (size_t)blockIdx.x * blockDim.x + threadIdx.x; // ((b*SEQ+n)*16+h)*64+d
  int d = (int)(idx & 63);
  int h = (int)((idx >> 6) & 15);
  size_t bn = idx >> 10;            // b*SEQ + n
  int n = (int)(bn & (SEQ - 1));
  int b = (int)(bn >> 11);
  size_t src = bn * 3072 + h * 64 + d;
  int bh = b * 16 + h;
  Q [((size_t)bh * SEQ + n) * 64 + d]  = qkv[src];
  Km[((size_t)bh * SEQ + n) * 64 + d]  = qkv[src + 1024];
  Vt[((size_t)bh * 64 + d) * SEQ + n]  = qkv[src + 2048];
}

// ---------------------------------------------------------------------------
// Causal flash attention: 1 wave per 16-query tile per (b,h).
// gridDim = (SEQ/16, B*H), blockDim = 32.
// ---------------------------------------------------------------------------
__global__ void __launch_bounds__(32, 1)
k_flash_attn(const bf16_t* __restrict__ Q, const bf16_t* __restrict__ K,
             const bf16_t* __restrict__ Vt, bf16_t* __restrict__ Out) {
  const int SEQ = 2048;
  __shared__ bf16_t Plds[16 * 32];

  const int lane = threadIdx.x & 31;
  const int half = lane >> 4;
  const int l16  = lane & 15;
  const int q0   = blockIdx.x * 16;
  const int bh   = blockIdx.y;

  const uint32_t qkbase = (uint32_t)bh * (uint32_t)(SEQ * 64); // elements
  const uint32_t vbase  = qkbase;

  v16bf qf[2];
#pragma unroll
  for (int kk = 0; kk < 2; ++kk)
    qf[kk] = load_fragA(Q, qkbase + (uint32_t)(q0 + l16) * 64u + kk * 32 + half * 16);

  v8f oacc[4];
  float rowmax[8], rowsum[8];
#pragma unroll
  for (int t = 0; t < 4; ++t)
#pragma unroll
    for (int r = 0; r < 8; ++r) oacc[t][r] = 0.0f;
#pragma unroll
  for (int r = 0; r < 8; ++r) { rowmax[r] = -3.0e38f; rowsum[r] = 0.0f; }

  const float scale = 0.125f; // 1/sqrt(64)
  const int jmax = (q0 + 15) >> 5;

  // prologue: K fragments for key block 0
  v16bf kf[2][2];
#pragma unroll
  for (int kk = 0; kk < 2; ++kk) {
    kf[kk][0] = load_fragB(K, qkbase + (uint32_t)(l16)      * 64u + kk * 32 + half * 16);
    kf[kk][1] = load_fragB(K, qkbase + (uint32_t)(16 + l16) * 64u + kk * 32 + half * 16);
  }

  for (int j = 0; j <= jmax; ++j) {
    const int key0 = j * 32;

    v8f s0, s1;
#pragma unroll
    for (int r = 0; r < 8; ++r) { s0[r] = 0.0f; s1[r] = 0.0f; }
#pragma unroll
    for (int kk = 0; kk < 2; ++kk) {
      s0 = wmma_bf16(qf[kk], kf[kk][0], s0);
      s1 = wmma_bf16(qf[kk], kf[kk][1], s1);
    }

    // V loads for this block + K loads for the next block issued here so
    // they overlap the softmax VALU below.
    v16bf vf[4];
#pragma unroll
    for (int t = 0; t < 4; ++t)
      vf[t] = load_fragB(Vt, vbase + (uint32_t)(t * 16 + l16) * (uint32_t)SEQ +
                              (uint32_t)key0 + half * 16);

    const int keyn = (j < jmax) ? (key0 + 32) : key0; // dummy reload at end
#pragma unroll
    for (int kk = 0; kk < 2; ++kk) {
      kf[kk][0] = load_fragB(K, qkbase + (uint32_t)(keyn + l16)      * 64u + kk * 32 + half * 16);
      kf[kk][1] = load_fragB(K, qkbase + (uint32_t)(keyn + 16 + l16) * 64u + kk * 32 + half * 16);
    }

#pragma unroll
    for (int r = 0; r < 8; ++r) {
      const int qrow = q0 + r + 8 * half;
      float v0 = s0[r] * scale;
      float v1 = s1[r] * scale;
      if (key0 + l16 > qrow)      v0 = -3.0e38f;
      if (key0 + 16 + l16 > qrow) v1 = -3.0e38f;

      float m = fmaxf(v0, v1);
#pragma unroll
      for (int sh = 8; sh >= 1; sh >>= 1) m = fmaxf(m, __shfl_xor(m, sh, 32));
      const float newmax = fmaxf(rowmax[r], m);
      const float alpha  = __expf(rowmax[r] - newmax);
      const float e0 = __expf(v0 - newmax);
      const float e1 = __expf(v1 - newmax);
      float ps = e0 + e1;
#pragma unroll
      for (int sh = 8; sh >= 1; sh >>= 1) ps += __shfl_xor(ps, sh, 32);
      rowsum[r] = rowsum[r] * alpha + ps;
      rowmax[r] = newmax;
#pragma unroll
      for (int t = 0; t < 4; ++t) oacc[t][r] *= alpha;

      Plds[(r + 8 * half) * 32 + l16]      = (bf16_t)e0;
      Plds[(r + 8 * half) * 32 + 16 + l16] = (bf16_t)e1;
    }
    __syncthreads();

    v16bf pf = load_fragA_lds(&Plds[l16 * 32 + half * 16]);
#pragma unroll
    for (int t = 0; t < 4; ++t)
      oacc[t] = wmma_bf16(pf, vf[t], oacc[t]);
    __syncthreads();
  }

  // write to [b, n, h*64 + d] bf16
  const int b = bh >> 4, h = bh & 15;
#pragma unroll
  for (int t = 0; t < 4; ++t) {
#pragma unroll
    for (int r = 0; r < 8; ++r) {
      const int qrow = q0 + r + 8 * half;
      const int d = t * 16 + l16;
      const float v = oacc[t][r] / rowsum[r];
      Out[((size_t)(b * SEQ + qrow)) * 1024 + h * 64 + d] = (bf16_t)v;
    }
  }
}

// ---------------------------------------------------------------------------
// Host-side launch
// ---------------------------------------------------------------------------
extern "C" void kernel_launch(void* const* d_in, const int* in_sizes, int n_in,
                              void* d_out, int out_size, void* d_ws, size_t ws_size,
                              hipStream_t stream) {
  const float* x      = (const float*)d_in[0]; // [4,2048,1024]
  const float* w_qkv  = (const float*)d_in[1]; // [1024,3072]
  const float* w_out  = (const float*)d_in[2]; // [1024,1024]
  const float* b_out  = (const float*)d_in[3]; // [1024]
  float* out = (float*)d_out;

  const int B = 4, SEQ = 2048, DM = 1024, INNER = 1024;
  const int M = B * SEQ; // 8192

  uint8_t* ws = (uint8_t*)d_ws;
  size_t off = 0;
  auto alloc = [&](size_t bytes) -> void* {
    void* p = ws + off;
    off += (bytes + 255) & ~(size_t)255;
    return p;
  };
  bf16_t* xbf   = (bf16_t*)alloc((size_t)M * DM * 2);
  bf16_t* wqkvT = (bf16_t*)alloc((size_t)3 * INNER * DM * 2);
  bf16_t* woutT = (bf16_t*)alloc((size_t)DM * INNER * 2);
  bf16_t* qkvb  = (bf16_t*)alloc((size_t)M * 3 * INNER * 2);
  bf16_t* Qb    = (bf16_t*)alloc((size_t)M * INNER * 2);
  bf16_t* Kb    = (bf16_t*)alloc((size_t)M * INNER * 2);
  bf16_t* Vtb   = (bf16_t*)alloc((size_t)M * INNER * 2);
  bf16_t* attn  = (bf16_t*)alloc((size_t)M * INNER * 2);

  // 1) fp32 -> bf16 conversions / weight transposes
  {
    size_t n = (size_t)M * DM;
    k_f32_to_bf16<<<(unsigned)((n + 255) / 256), 256, 0, stream>>>(x, xbf, n);
  }
  {
    size_t n = (size_t)DM * 3 * INNER;
    k_transpose_bf16<<<(unsigned)((n + 255) / 256), 256, 0, stream>>>(w_qkv, wqkvT, DM, 3 * INNER);
  }
  {
    size_t n = (size_t)INNER * DM;
    k_transpose_bf16<<<(unsigned)((n + 255) / 256), 256, 0, stream>>>(w_out, woutT, INNER, DM);
  }

  // 2) QKV projection: qkvb[M, 3072] = xbf[M,1024] * wqkvT^T  (bf16 out)
  {
    dim3 grid(3 * INNER / 64, M / 256);
    k_gemm_bf16<<<grid, 256, 0, stream>>>(xbf, wqkvT, nullptr, qkvb, nullptr,
                                          M, 3 * INNER, DM);
  }

  // 3) repack into attention-friendly layouts
  {
    size_t n = (size_t)M * INNER;
    k_repack_qkv<<<(unsigned)(n / 256), 256, 0, stream>>>(qkvb, Qb, Kb, Vtb);
  }

  // 4) causal flash attention
  {
    dim3 grid(SEQ / 16, B * 16);
    k_flash_attn<<<grid, 32, 0, stream>>>(Qb, Kb, Vtb, attn);
  }

  // 5) output projection + bias: out[M,1024] = attn[M,1024] * woutT^T + b_out
  {
    dim3 grid(DM / 64, M / 256);
    k_gemm_bf16<<<grid, 256, 0, stream>>>(attn, woutT, out, nullptr, b_out,
                                          M, DM, INNER);
  }
}
